// GraphTransformerWithPositionalEncoding_28303834481183
// MI455X (gfx1250) — matrix operand order repaired and checked by reading, not compiled, not run on hardware
//
#include <hip/hip_runtime.h>
#include <math.h>

typedef __attribute__((ext_vector_type(16))) _Float16 v16h;
typedef __attribute__((ext_vector_type(8)))  float    v8f;

// ---------------------------------------------------------------- utilities

__device__ __forceinline__ float wave_reduce_sum(float v) {
#pragma unroll
  for (int off = 16; off > 0; off >>= 1)
    v += __shfl_xor(v, off, 32);
  return v;
}

__device__ __forceinline__ void atomic_max_float(float* addr, float val) {
  // Sign-aware int-punning float max (valid when addr initialized to -inf).
  if (val >= 0.0f)
    atomicMax((int*)addr, __float_as_int(val));
  else
    atomicMin((unsigned int*)addr, __float_as_uint(val));
}

__global__ void fill_kernel(float* __restrict__ p, float val, int n) {
  int i = blockIdx.x * blockDim.x + threadIdx.x;
  if (i < n) p[i] = val;
}

// --------------------- weight prep: W[din x dout] f32 -> Wt[dout x din] f16

__global__ void transpose_f16_kernel(const float* __restrict__ w,
                                     _Float16* __restrict__ wt,
                                     int din, int dout) {
  int idx = blockIdx.x * blockDim.x + threadIdx.x;
  if (idx >= din * dout) return;
  int o = idx / din;
  int k = idx - o * din;
  wt[idx] = (_Float16)w[(size_t)k * dout + o];  // contiguous write, strided read
}

// ------------------------------------------- stage 0: h0 = [x | pos@W + b]

__global__ void pos_encode_kernel(const float* __restrict__ x,
                                  const float* __restrict__ pos,
                                  const float* __restrict__ wpos,
                                  const float* __restrict__ bpos,
                                  float* __restrict__ h0, int n) {
  int idx = blockIdx.x * blockDim.x + threadIdx.x;
  if (idx >= n * 64) return;
  int i = idx >> 6;   // node
  int j = idx & 63;   // feature
  float v;
  if (j < 32) {
    v = x[(size_t)i * 32 + j];
  } else {
    int c = j - 32;
    const float* p = pos + (size_t)i * 3;
    v = p[0] * wpos[0 * 32 + c] + p[1] * wpos[1 * 32 + c] +
        p[2] * wpos[2 * 32 + c] + bpos[c];
  }
  h0[idx] = v;
}

// ------------------------------------------- WMMA GEMM: out = A @ W + bias
// A: [rows x din] f32 row-major (rows % 16 == 0). Wt: [dout x din] f16
// (pre-transposed). One wave owns a 16-row strip and ALL TN = dout/16 output
// tiles: A fragment loaded once per 32-wide K-chunk and reused for TN
// back-to-back v_wmma_f32_16x16x32_f16; B fragment is one aligned 32-byte
// v16h load per tile.

template <int TN>
__global__ __launch_bounds__(256) void gemm_wmma_kernel(
    const float* __restrict__ A, const _Float16* __restrict__ Wt,
    const float* __restrict__ bias, float* __restrict__ out,
    int rows, int din) {
  constexpr int dout = TN * 16;
  int wave   = blockIdx.x * (blockDim.x >> 5) + (threadIdx.x >> 5);
  int tilesM = rows >> 4;
  if (wave >= tilesM) return;            // wave-uniform: EXEC all-ones below
  int lane = threadIdx.x & 31;
  int l16  = lane & 15;
  int hs   = lane >> 4;                  // half-wave select
  const float* arow = A + (size_t)(wave * 16 + l16) * din;

  v8f acc[TN] = {};
  for (int kk = 0; kk < din; kk += 32) {
    // A 16x32 f16 layout: lanes<16 -> K 0..15 ; lanes>=16 -> K 16..31 (matches B)
    v16h a;
    int ka = kk + hs * 16;
#pragma unroll
    for (int i = 0; i < 8; ++i) {
      a[i]     = (_Float16)arow[ka + i];
      a[8 + i] = (_Float16)arow[ka + 8 + i];
    }
    // B 32x16 f16 layout: lanes<16 -> N=lane, K 0..15 ; lanes>=16 -> K 16..31.
    // With Wt[dout][din], a lane's 16 halves are contiguous & 32B-aligned.
    int kb = kk + hs * 16;
#pragma unroll
    for (int t = 0; t < TN; ++t) {
      int col = t * 16 + l16;
      v16h b  = *(const v16h*)(Wt + (size_t)col * din + kb);
      acc[t] = __builtin_amdgcn_wmma_f32_16x16x32_f16(
          /*neg_a=*/false, a, /*neg_b=*/false, b,
          /*c_mod=*/(short)0, acc[t], /*reuse_a=*/false, /*reuse_b=*/false);
    }
  }

#pragma unroll
  for (int t = 0; t < TN; ++t) {
    int col  = t * 16 + l16;
    float bv = bias[col];
#pragma unroll
    for (int r = 0; r < 8; ++r) {
      int mrow = wave * 16 + r + hs * 8;  // C layout: VGPR r -> M=r / M=r+8
      out[(size_t)mrow * dout + col] = acc[t][r] + bv;
    }
  }
}

// ------------------------------------------- edge pass 1: logits + seg-max
// One wave per (edge, head); lane = channel (ch == 32).

__global__ void edge_logits_kernel(const float* __restrict__ q,
                                   const float* __restrict__ k,
                                   const int* __restrict__ src,
                                   const int* __restrict__ dst,
                                   float* __restrict__ lg,
                                   float* __restrict__ m,
                                   int E_, int heads, float scale) {
  int wave = (blockIdx.x * blockDim.x + threadIdx.x) >> 5;
  if (wave >= E_ * heads) return;
  int lane = threadIdx.x & 31;
  int e = wave / heads, h = wave % heads;
  int s = src[e], d = dst[e];
  float p = q[((size_t)d * heads + h) * 32 + lane] *
            k[((size_t)s * heads + h) * 32 + lane];
  float sum = wave_reduce_sum(p) * scale;
  if (lane == 0) {
    lg[(size_t)e * heads + h] = sum;
    atomic_max_float(&m[(size_t)d * heads + h], sum);
  }
}

// --------------------------- edge pass 2: unnormalized softmax accumulation
// att[dst] += exp(l - m[dst]) * v[src] ; denom[dst] += exp(l - m[dst])

__global__ void edge_accum_kernel(const float* __restrict__ lg,
                                  const float* __restrict__ m,
                                  const float* __restrict__ v,
                                  const int* __restrict__ src,
                                  const int* __restrict__ dst,
                                  float* __restrict__ att,
                                  float* __restrict__ denom,
                                  int E_, int heads) {
  int wave = (blockIdx.x * blockDim.x + threadIdx.x) >> 5;
  if (wave >= E_ * heads) return;
  int lane = threadIdx.x & 31;
  int e = wave / heads, h = wave % heads;
  int s = src[e], d = dst[e];
  float a = __expf(lg[(size_t)e * heads + h] - m[(size_t)d * heads + h]);
  atomicAdd(&att[((size_t)d * heads + h) * 32 + lane],
            a * v[((size_t)s * heads + h) * 32 + lane]);
  if (lane == 0) atomicAdd(&denom[(size_t)d * heads + h], a);
}

// ---------------------- node pass: out = skip + att/denom (optional ReLU)

__global__ void node_finish_kernel(const float* __restrict__ att,
                                   const float* __restrict__ denom,
                                   float* __restrict__ out,
                                   int n, int heads, int relu) {
  int feat = heads * 32;
  int idx  = blockIdx.x * blockDim.x + threadIdx.x;
  if (idx >= n * feat) return;
  int i = idx / feat;
  int h = (idx % feat) >> 5;
  float dn = denom[(size_t)i * heads + h];
  float a  = (dn > 0.0f) ? att[idx] / dn : 0.0f;  // empty segment -> 0
  float v  = out[idx] + a;
  if (relu) v = fmaxf(v, 0.0f);
  out[idx] = v;
}

// ---------------------------------------------------- LayerNorm (32 chans)

__global__ void layer_norm_kernel(const float* __restrict__ in,
                                  const float* __restrict__ g,
                                  const float* __restrict__ b,
                                  float* __restrict__ out, int n) {
  int wave = (blockIdx.x * blockDim.x + threadIdx.x) >> 5;
  if (wave >= n) return;
  int lane = threadIdx.x & 31;
  float v   = in[(size_t)wave * 32 + lane];
  float mu  = wave_reduce_sum(v) * (1.0f / 32.0f);
  float dv  = v - mu;
  float var = wave_reduce_sum(dv * dv) * (1.0f / 32.0f);
  out[(size_t)wave * 32 + lane] = dv * rsqrtf(var + 1e-5f) * g[lane] + b[lane];
}

// ------------------------------------------------------------------- host

extern "C" void kernel_launch(void* const* d_in, const int* in_sizes, int n_in,
                              void* d_out, int out_size, void* d_ws, size_t ws_size,
                              hipStream_t stream) {
  const float* x    = (const float*)d_in[0];
  const float* pos  = (const float*)d_in[1];
  const int*   edge = (const int*)d_in[2];
  const float* wpos = (const float*)d_in[3];
  const float* bpos = (const float*)d_in[4];
  const float* wq0 = (const float*)d_in[5],  *bq0 = (const float*)d_in[6];
  const float* wk0 = (const float*)d_in[7],  *bk0 = (const float*)d_in[8];
  const float* wv0 = (const float*)d_in[9],  *bv0 = (const float*)d_in[10];
  const float* ws0 = (const float*)d_in[11], *bs0 = (const float*)d_in[12];
  const float* wq1 = (const float*)d_in[13], *bq1 = (const float*)d_in[14];
  const float* wk1 = (const float*)d_in[15], *bk1 = (const float*)d_in[16];
  const float* wv1 = (const float*)d_in[17], *bv1 = (const float*)d_in[18];
  const float* ws1 = (const float*)d_in[19], *bs1 = (const float*)d_in[20];
  const float* wq2 = (const float*)d_in[21], *bq2 = (const float*)d_in[22];
  const float* wk2 = (const float*)d_in[23], *bk2 = (const float*)d_in[24];
  const float* wv2 = (const float*)d_in[25], *bv2 = (const float*)d_in[26];
  const float* ws2 = (const float*)d_in[27], *bs2 = (const float*)d_in[28];
  const float* lng = (const float*)d_in[29], *lnb = (const float*)d_in[30];

  const int N = in_sizes[0] / 32;
  const int E = in_sizes[2] / 2;
  const int* src = edge;
  const int* dst = edge + E;
  const float scale = 0.17677669529663687f;  // 1/sqrt(32)

  float* ws = (float*)d_ws;
  size_t o = 0;
  float* P0 = ws + o; o += (size_t)N * 128;  // layer input / output ping
  float* P1 = ws + o; o += (size_t)N * 128;  // layer input / output pong
  float* Q  = ws + o; o += (size_t)N * 128;
  float* K  = ws + o; o += (size_t)N * 128;  // also reused as att accumulator
  float* V  = ws + o; o += (size_t)N * 128;
  float* M  = ws + o; o += (size_t)N * 4;
  float* D  = ws + o; o += (size_t)N * 4;
  float* LG = ws + o; o += (size_t)E * 4;
  o = (o + 15) & ~(size_t)15;                // 64B-align f16 weight pool
  _Float16* wt_pool = (_Float16*)(ws + o);
  (void)ws_size; (void)n_in; (void)out_size;

  // ---- one-time: transpose+convert all 12 weight matrices to f16 [dout][din]
  _Float16* wt[12];
  const float* wsrc[12] = {wq0, wk0, wv0, ws0, wq1, wk1, wv1, ws1,
                           wq2, wk2, wv2, ws2};
  const int   wdin[12]  = {64, 64, 64, 64, 128, 128, 128, 128,
                           128, 128, 128, 128};
  const int   wdout[12] = {128, 128, 128, 128, 128, 128, 128, 128,
                           32, 32, 32, 32};
  {
    size_t wo = 0;
    for (int i = 0; i < 12; ++i) {
      wt[i] = wt_pool + wo;
      int ne = wdin[i] * wdout[i];
      wo += (size_t)ne;
      transpose_f16_kernel<<<(ne + 255) / 256, 256, 0, stream>>>(
          wsrc[i], wt[i], wdin[i], wdout[i]);
    }
  }

  auto fills = [&](float* p, float v, size_t n) {
    fill_kernel<<<(int)((n + 255) / 256), 256, 0, stream>>>(p, v, (int)n);
  };

  int gemm_blocks = ((N >> 4) + 7) / 8;       // 1 wave per 16-row strip

  auto run_layer = [&](const float* hin, int din, int heads, int wbase,
                       const float* bq, const float* bk, const float* bv,
                       const float* bsk, float* hout, int relu) {
    int dout = heads * 32;
    if (dout == 128) {
      gemm_wmma_kernel<8><<<gemm_blocks, 256, 0, stream>>>(hin, wt[wbase + 0], bq,  Q,    N, din);
      gemm_wmma_kernel<8><<<gemm_blocks, 256, 0, stream>>>(hin, wt[wbase + 1], bk,  K,    N, din);
      gemm_wmma_kernel<8><<<gemm_blocks, 256, 0, stream>>>(hin, wt[wbase + 2], bv,  V,    N, din);
      gemm_wmma_kernel<8><<<gemm_blocks, 256, 0, stream>>>(hin, wt[wbase + 3], bsk, hout, N, din);
    } else {
      gemm_wmma_kernel<2><<<gemm_blocks, 256, 0, stream>>>(hin, wt[wbase + 0], bq,  Q,    N, din);
      gemm_wmma_kernel<2><<<gemm_blocks, 256, 0, stream>>>(hin, wt[wbase + 1], bk,  K,    N, din);
      gemm_wmma_kernel<2><<<gemm_blocks, 256, 0, stream>>>(hin, wt[wbase + 2], bv,  V,    N, din);
      gemm_wmma_kernel<2><<<gemm_blocks, 256, 0, stream>>>(hin, wt[wbase + 3], bsk, hout, N, din);
    }
    fills(M, -INFINITY, (size_t)N * heads);
    fills(D, 0.0f,      (size_t)N * heads);
    int ewaves = E * heads;
    int egb    = (ewaves + 7) / 8;
    edge_logits_kernel<<<egb, 256, 0, stream>>>(Q, K, src, dst, LG, M, E, heads, scale);
    fills(K, 0.0f, (size_t)N * dout);          // recycle K as attention accum
    edge_accum_kernel<<<egb, 256, 0, stream>>>(LG, M, V, src, dst, K, D, E, heads);
    int nf = N * dout;
    node_finish_kernel<<<(nf + 255) / 256, 256, 0, stream>>>(K, D, hout, N, heads, relu);
  };

  // stage 0: h0 = [x | pos @ w_pos + b_pos]  -> P0 [N,64]
  pos_encode_kernel<<<(N * 64 + 255) / 256, 256, 0, stream>>>(x, pos, wpos, bpos, P0, N);

  run_layer(P0,  64, 4, 0, bq0, bk0, bv0, bs0, P1, 1);
  run_layer(P1, 128, 4, 4, bq1, bk1, bv1, bs1, P0, 1);
  run_layer(P0, 128, 1, 8, bq2, bk2, bv2, bs2, P1, 0);

  layer_norm_kernel<<<(N + 7) / 8, 256, 0, stream>>>(P1, lng, lnb, (float*)d_out, N);
}
